// AttentionHead_52707838657300
// MI455X (gfx1250) — matrix operand order repaired and checked by reading, not compiled, not run on hardware
//
#include <hip/hip_runtime.h>

// ---------------------------------------------------------------------------
// Attention head (RoPE) for MI455X / gfx1250, wave32 + v_wmma_f32_16x16x32_f16
// B=8, T=2048, C=1024, H=128.  Pipeline:
//   0) wconv_kernel: Wq/Wk/Wv f32 -> f16 once (kills per-block re-conversion)
//   1) proj_rope_kernel: q/k/v = x@W (f16 WMMA, f32 acc), RoPE on q,k.
//      Q  -> [B*T][H]  f16   (row-major: A-operand layout for Q@K^T)
//      Kt -> [B][H][T] f16   (transposed: B-operand rows are contiguous)
//      V  -> [B*T][H]  f16   (B-operand rows for P@V are contiguous)
//   2) attn_kernel: flash-style causal softmax(Q@K^T)@V, one wave per
//      16-query tile, online softmax in fp32, next-chunk prefetch.
// ---------------------------------------------------------------------------

typedef __attribute__((ext_vector_type(16))) _Float16 v16h;
typedef __attribute__((ext_vector_type(8)))  float    v8f;

#define Bdim 8
#define Tdim 2048
#define Cdim 1024
#define Hdim 128

// Load a 16x32 f16 A-operand tile from a row-major matrix (row stride `ld`
// elements).  Per ISA 7.12.2 (16-bit A 16x32): lane L holds row M=L%16;
// VGPR j (halves 2j,2j+1): K = 8*(L/16) + 2j            for j=0..3
//                          K = 16 + 8*(L/16) + 2*(j-4)  for j=4..7
__device__ inline v16h load_a_f16(const _Float16* base, int ld, int M, int hi) {
    union { v16h v; unsigned u[8]; } r;
    const _Float16* row = base + (size_t)M * ld;
    const int k0 = hi * 8;
#pragma unroll
    for (int j = 0; j < 4; ++j)
        r.u[j] = *(const unsigned*)(row + k0 + 2 * j);
#pragma unroll
    for (int j = 0; j < 4; ++j)
        r.u[4 + j] = *(const unsigned*)(row + 16 + k0 + 2 * j);
    return r.v;
}

// One-shot f32 -> f16 weight conversion ([Cdim][Hdim] = 32768 float4s).
__global__ __launch_bounds__(256) void wconv_kernel(
    const float* __restrict__ W, _Float16* __restrict__ W16) {
    int idx = blockIdx.x * 256 + threadIdx.x;      // float4 index
    float4 f = ((const float4*)W)[idx];
    union { _Float16 h[4]; uint2 u2; } pk;
    pk.h[0] = (_Float16)f.x; pk.h[1] = (_Float16)f.y;
    pk.h[2] = (_Float16)f.z; pk.h[3] = (_Float16)f.w;
    *(uint2*)(W16 + (size_t)idx * 4) = pk.u2;
}

__global__ __launch_bounds__(256) void proj_rope_kernel(
    const float*    __restrict__ x,
    const _Float16* __restrict__ Wq16, const _Float16* __restrict__ Wk16,
    const _Float16* __restrict__ Wv16,
    _Float16* __restrict__ Q, _Float16* __restrict__ Kt,
    _Float16* __restrict__ V) {
    __shared__ _Float16 xbuf[16 * Cdim];   // 32 KB: x tile as f16
    __shared__ float    qbuf[16 * Hdim];   //  8 KB
    __shared__ float    kbuf[16 * Hdim];   //  8 KB

    const int tid = threadIdx.x;
    const int r0  = blockIdx.x * 16;       // token tile base, flattened B*T

    // Stage 16x1024 x tile into LDS as f16 (tile rows are contiguous).
    const float4* x4 = (const float4*)(x + (size_t)r0 * Cdim);
#pragma unroll
    for (int e = 0; e < 16; ++e) {
        int idx = e * 256 + tid;           // 0..4095 float4s
        float4 f = x4[idx];
        union { _Float16 h[4]; uint2 u2; } pk;
        pk.h[0] = (_Float16)f.x; pk.h[1] = (_Float16)f.y;
        pk.h[2] = (_Float16)f.z; pk.h[3] = (_Float16)f.w;
        *(uint2*)(&xbuf[idx * 4]) = pk.u2;
    }
    __syncthreads();

    const int wv   = tid >> 5;             // wave id = output h-tile
    const int lane = tid & 31;
    const int M    = lane & 15;
    const int hi   = lane >> 4;
    const int h0   = wv * 16;

    v8f aq = {0.f,0.f,0.f,0.f,0.f,0.f,0.f,0.f};
    v8f ak = aq, av = aq;

    for (int kk = 0; kk < Cdim; kk += 32) {
        v16h a = load_a_f16(xbuf + kk, Cdim, M, hi);
        // B operand: lane = N column (h0+M), elements over K in table order, f16.
        v16h bq, bk, bv;
#pragma unroll
        for (int s = 0; s < 16; ++s) {
            const int ks = (s < 8) ? (8 * hi + s) : (16 + 8 * hi + s - 8);
            const size_t wro = (size_t)(kk + ks) * Hdim + h0 + M;
            bq[s] = Wq16[wro];
            bk[s] = Wk16[wro];
            bv[s] = Wv16[wro];
        }
        aq = __builtin_amdgcn_wmma_f32_16x16x32_f16(false, a, false, bq,
                                                    (short)0, aq, false, false);
        ak = __builtin_amdgcn_wmma_f32_16x16x32_f16(false, a, false, bk,
                                                    (short)0, ak, false, false);
        av = __builtin_amdgcn_wmma_f32_16x16x32_f16(false, a, false, bv,
                                                    (short)0, av, false, false);
    }

    // V: straight f16 store from C-layout (row = v + 8*hi, col = h0 + M).
#pragma unroll
    for (int v = 0; v < 8; ++v) {
        int t_loc = v + 8 * hi;
        V[(size_t)(r0 + t_loc) * Hdim + h0 + M] = (_Float16)av[v];
    }
    // q,k: stage fp32 in LDS so RoPE can pair h with h +/- 64.
#pragma unroll
    for (int v = 0; v < 8; ++v) {
        int t_loc = v + 8 * hi;
        qbuf[t_loc * Hdim + h0 + M] = aq[v];
        kbuf[t_loc * Hdim + h0 + M] = ak[v];
    }
    __syncthreads();

    const float LOG_BASE = 9.210340371976184f;  // ln(10000)
#pragma unroll
    for (int e = 0; e < 8; ++e) {
        int idx   = e * 256 + tid;         // 0..2047 -> (t_loc, h)
        int t_loc = idx >> 7;
        int h     = idx & 127;
        int g     = r0 + t_loc;
        int bb    = g >> 11;               // / T
        int tt    = g & (Tdim - 1);        // % T
        float inv  = __expf(-(float)(h & 63) * (2.0f / 128.0f) * LOG_BASE);
        float freq = (float)tt * inv;
        float cs = __cosf(freq), sn = __sinf(freq);
        float sgn = (h < 64) ? -1.0f : 1.0f;
        int   ho  = (h + 64) & 127;
        float qr = qbuf[t_loc * Hdim + h] * cs + sgn * qbuf[t_loc * Hdim + ho] * sn;
        float kr = kbuf[t_loc * Hdim + h] * cs + sgn * kbuf[t_loc * Hdim + ho] * sn;
        Q[(size_t)g * Hdim + h] = (_Float16)qr;
        Kt[((size_t)bb * Hdim + h) * Tdim + tt] = (_Float16)kr;
    }
}

__global__ __launch_bounds__(32) void attn_kernel(
    const _Float16* __restrict__ Q, const _Float16* __restrict__ Kt,
    const _Float16* __restrict__ V, float* __restrict__ out) {
    __shared__ _Float16 pbuf[16 * 32];     // C-layout -> A-layout transpose

    const int lane = threadIdx.x & 31;
    const int qi   = blockIdx.x;           // query tile in T
    const int b    = blockIdx.y;
    const int t0   = qi * 16;
    const int M    = lane & 15;
    const int hi   = lane >> 4;
    const int col  = M;

    // Q tile 16x128 in A-operand layout (4 chunks of K=32).
    const _Float16* Qb = Q + ((size_t)b * Tdim + t0) * Hdim;
    v16h qa[4];
#pragma unroll
    for (int c = 0; c < 4; ++c) qa[c] = load_a_f16(Qb + c * 32, Hdim, M, hi);

    v8f  acc[8];
    float mrow[8], lrow[8];
    v8f zf = {0.f,0.f,0.f,0.f,0.f,0.f,0.f,0.f};
#pragma unroll
    for (int n = 0; n < 8; ++n) acc[n] = zf;
#pragma unroll
    for (int v = 0; v < 8; ++v) { mrow[v] = -1e30f; lrow[v] = 0.0f; }

    const _Float16* Ktb = Kt + (size_t)b * Hdim * Tdim;
    const _Float16* Vb  = V  + (size_t)b * Tdim * Hdim;
    const int nch = (qi >> 1) + 1;         // 32-key chunks up to the diagonal

    for (int cj = 0; cj < nch; ++cj) {
        const int s0 = cj * 32;
        // Prefetch next chunk's K/V cachelines (global_prefetch_b8).
        if (cj + 1 < nch) {
            const int sn = s0 + 32;
#pragma unroll
            for (int hc = 0; hc < 4; ++hc)
                __builtin_prefetch(Ktb + (size_t)(hc * 32 + lane) * Tdim + sn, 0, 1);
            __builtin_prefetch(Vb + (size_t)(sn + lane) * Hdim, 0, 1);
        }
        // S = Q K^T : two 16x16 tiles (s columns 0-15 / 16-31 of the chunk)
        v8f sA = zf, sB = zf;
#pragma unroll
        for (int hc = 0; hc < 4; ++hc) {
            v16h b0, b1;
#pragma unroll
            for (int s = 0; s < 16; ++s) {
                const int ks = (s < 8) ? (8 * hi + s) : (16 + 8 * hi + s - 8);
                const _Float16* krow = Ktb + (size_t)(hc * 32 + ks) * Tdim + s0 + M;
                b0[s] = krow[0];
                b1[s] = krow[16];
            }
            sA = __builtin_amdgcn_wmma_f32_16x16x32_f16(false, qa[hc], false, b0,
                                                        (short)0, sA, false, false);
            sB = __builtin_amdgcn_wmma_f32_16x16x32_f16(false, qa[hc], false, b1,
                                                        (short)0, sB, false, false);
        }
        // Online softmax (C-layout: VGPR v holds row v + 8*hi, col = lane%16).
#pragma unroll
        for (int v = 0; v < 8; ++v) {
            int   tg = t0 + v + 8 * hi;
            float e0 = sA[v], e1 = sB[v];
            if (s0 + col > tg)      e0 = -1e30f;   // causal mask
            if (s0 + 16 + col > tg) e1 = -1e30f;
            float pm = fmaxf(e0, e1);
            pm = fmaxf(pm, __shfl_xor(pm, 1));
            pm = fmaxf(pm, __shfl_xor(pm, 2));
            pm = fmaxf(pm, __shfl_xor(pm, 4));
            pm = fmaxf(pm, __shfl_xor(pm, 8));
            float mnew = fmaxf(mrow[v], pm);
            float sc   = __expf(mrow[v] - mnew);
            mrow[v] = mnew;
            float p0 = __expf(e0 - mnew), p1 = __expf(e1 - mnew);
            float ps = p0 + p1;
            ps += __shfl_xor(ps, 1);
            ps += __shfl_xor(ps, 2);
            ps += __shfl_xor(ps, 4);
            ps += __shfl_xor(ps, 8);
            lrow[v] = lrow[v] * sc + ps;
#pragma unroll
            for (int n = 0; n < 8; ++n) acc[n][v] *= sc;
            pbuf[(v + 8 * hi) * 32 + col]      = (_Float16)p0;
            pbuf[(v + 8 * hi) * 32 + 16 + col] = (_Float16)p1;
        }
        __syncthreads();                       // order LDS within the wave
        v16h pa = load_a_f16(pbuf, 32, M, hi); // P in A-operand layout
        __syncthreads();
        // O += P V  (B operand: lane = h column, elements over s in table order)
#pragma unroll
        for (int n = 0; n < 8; ++n) {
            v16h bv;
#pragma unroll
            for (int s = 0; s < 16; ++s) {
                const int ks = (s < 8) ? (8 * hi + s) : (16 + 8 * hi + s - 8);
                bv[s] = Vb[(size_t)(s0 + ks) * Hdim + n * 16 + M];
            }
            acc[n] = __builtin_amdgcn_wmma_f32_16x16x32_f16(false, pa, false, bv,
                                                            (short)0, acc[n], false, false);
        }
    }

    float* ob = out + ((size_t)b * Tdim + t0) * Hdim;
#pragma unroll
    for (int v = 0; v < 8; ++v) {
        float il = 1.0f / lrow[v];
#pragma unroll
        for (int n = 0; n < 8; ++n)
            ob[(size_t)(v + 8 * hi) * Hdim + n * 16 + col] = acc[n][v] * il;
    }
}

extern "C" void kernel_launch(void* const* d_in, const int* in_sizes, int n_in,
                              void* d_out, int out_size, void* d_ws, size_t ws_size,
                              hipStream_t stream) {
    const float* x  = (const float*)d_in[0];
    const float* Wq = (const float*)d_in[1];
    const float* Wk = (const float*)d_in[2];
    const float* Wv = (const float*)d_in[3];
    float* out = (float*)d_out;

    // Workspace: Q (4MB) | Kt (4MB) | V (4MB) | Wq16/Wk16/Wv16 (256KB each).
    char* ws = (char*)d_ws;
    const size_t MB = 1024 * 1024;
    _Float16* Qh   = (_Float16*)(ws);
    _Float16* Kth  = (_Float16*)(ws + 4 * MB);
    _Float16* Vh   = (_Float16*)(ws + 8 * MB);
    _Float16* Wq16 = (_Float16*)(ws + 12 * MB);
    _Float16* Wk16 = (_Float16*)(ws + 12 * MB + 256 * 1024);
    _Float16* Wv16 = (_Float16*)(ws + 12 * MB + 512 * 1024);

    const int wblk = (Cdim * Hdim / 4) / 256;   // 128 blocks per weight matrix
    wconv_kernel<<<dim3(wblk), dim3(256), 0, stream>>>(Wq, Wq16);
    wconv_kernel<<<dim3(wblk), dim3(256), 0, stream>>>(Wk, Wk16);
    wconv_kernel<<<dim3(wblk), dim3(256), 0, stream>>>(Wv, Wv16);

    proj_rope_kernel<<<dim3((Bdim * Tdim) / 16), dim3(256), 0, stream>>>(
        x, Wq16, Wk16, Wv16, Qh, Kth, Vh);
    attn_kernel<<<dim3(Tdim / 16, Bdim), dim3(32), 0, stream>>>(
        Qh, Kth, Vh, out);
}
